// QCNN_60404420051309
// MI455X (gfx1250) — matrix operand order repaired and checked by reading, not compiled, not run on hardware
//
#include <hip/hip_runtime.h>
#include <math.h>

typedef __attribute__((ext_vector_type(8)))  _Float16 v8h;
typedef __attribute__((ext_vector_type(16))) _Float16 v16h;
typedef __attribute__((ext_vector_type(8)))  float    v8f;

#define B_DIM 128
#define L_DIM 4096
#define K_DIM 4096   /* padded OL (OL = 4095) */
#define OL    4095
#define NCLS  10

// ---------------------------------------------------------------------------
// Prep 1: G[b,o] = cos(in[b,o,1]) * cos(in[b,o+1,0]) * cos(in[b,o+1,1])
// stored as f16 hi + f16 lo planes (128 x 4096, k=4095 zero-padded).
// ---------------------------------------------------------------------------
__global__ void qcnn_prep_g(const float* __restrict__ in,
                            _Float16* __restrict__ ghi,
                            _Float16* __restrict__ glo) {
  int idx = blockIdx.x * blockDim.x + threadIdx.x;
  if (idx >= B_DIM * K_DIM) return;
  int b = idx >> 12;
  int o = idx & (K_DIM - 1);
  float g = 0.0f;
  if (o < OL) {
    const float* p = in + (size_t)b * (L_DIM * 2) + (size_t)o * 2;
    g = cosf(p[1]) * cosf(p[2]) * cosf(p[3]);
  }
  _Float16 hi = (_Float16)g;
  ghi[idx] = hi;
  glo[idx] = (_Float16)(g - (float)hi);
}

// ---------------------------------------------------------------------------
// Prep 2: W'^T[n,k] = C0*W[2k,n] + C1*W[2k+1,n], C_f = prod_{j=1..3} cos(p_fj)
// stored transposed (16 x 4096) as f16 hi + lo planes (n>=10 / k>=4095 -> 0).
// ---------------------------------------------------------------------------
__global__ void qcnn_prep_w(const float* __restrict__ W,
                            const float* __restrict__ params,
                            _Float16* __restrict__ whi,
                            _Float16* __restrict__ wlo) {
  int idx = blockIdx.x * blockDim.x + threadIdx.x;
  if (idx >= 16 * K_DIM) return;
  int n = idx >> 12;
  int k = idx & (K_DIM - 1);
  float v = 0.0f;
  if (n < NCLS && k < OL) {
    float c0 = cosf(params[1]) * cosf(params[2]) * cosf(params[3]);
    float c1 = cosf(params[5]) * cosf(params[6]) * cosf(params[7]);
    v = c0 * W[(size_t)(2 * k) * NCLS + n] +
        c1 * W[(size_t)(2 * k + 1) * NCLS + n];
  }
  _Float16 hi = (_Float16)v;
  whi[idx] = hi;
  wlo[idx] = (_Float16)(v - (float)hi);
}

// ---------------------------------------------------------------------------
// GEMM (128 x 4096) x (4096 x 16) via v_wmma_f32_16x16x32_f16, split-f16 for
// ~fp32 accuracy, split-K across 8 waves, LDS reduce, fused bias + softmax.
// Block t handles M rows [16t, 16t+16); grid = 8 blocks x 256 threads.
// ---------------------------------------------------------------------------
__global__ __launch_bounds__(256) void qcnn_gemm_softmax(
    const _Float16* __restrict__ Ghi, const _Float16* __restrict__ Glo,
    const _Float16* __restrict__ Whi, const _Float16* __restrict__ Wlo,
    const float* __restrict__ bias, float* __restrict__ out) {
  __shared__ float part[8][16][16];

  const int tid    = (int)threadIdx.x;
  const int wave   = tid >> 5;
  const int lane   = tid & 31;
  const int half   = lane >> 4;   // which 16-lane half of the wave
  const int lane15 = lane & 15;
  const int mtile  = (int)blockIdx.x;

  // A fragment: lane -> row M; halves hold {K+0..7, K+16..23} (lanes 0-15)
  //                               or    {K+8..15, K+24..31} (lanes 16-31)
  const size_t arow = (size_t)(mtile * 16 + lane15) * K_DIM + (half ? 8 : 0);
  // B fragment: lane -> col N; halves hold {K+0..7, K+16..23} (lanes 0-15) / {K+8..15, K+24..31}
  const size_t brow = (size_t)lane15 * K_DIM + (half ? 8 : 0);

  v8f c = {};
  const int kbase = wave * (K_DIM / 8);  // 512 per wave
  for (int k = 0; k < K_DIM / 8; k += 32) {
    const size_t ka = arow + (size_t)(kbase + k);
    const size_t kb = brow + (size_t)(kbase + k);
    v8h ah0 = *(const v8h*)(Ghi + ka);
    v8h ah1 = *(const v8h*)(Ghi + ka + 16);
    v8h al0 = *(const v8h*)(Glo + ka);
    v8h al1 = *(const v8h*)(Glo + ka + 16);
    v8h bh0 = *(const v8h*)(Whi + kb);
    v8h bh1 = *(const v8h*)(Whi + kb + 16);
    v8h bl0 = *(const v8h*)(Wlo + kb);
    v8h bl1 = *(const v8h*)(Wlo + kb + 16);
    v16h ah = __builtin_shufflevector(ah0, ah1, 0,1,2,3,4,5,6,7,8,9,10,11,12,13,14,15);
    v16h al = __builtin_shufflevector(al0, al1, 0,1,2,3,4,5,6,7,8,9,10,11,12,13,14,15);
    v16h bh = __builtin_shufflevector(bh0, bh1, 0,1,2,3,4,5,6,7,8,9,10,11,12,13,14,15);
    v16h bl = __builtin_shufflevector(bl0, bl1, 0,1,2,3,4,5,6,7,8,9,10,11,12,13,14,15);
    // D = Ah*Bh + Ah*Bl + Al*Bh + C  (Al*Bl term ~2^-22, dropped)
    c = __builtin_amdgcn_wmma_f32_16x16x32_f16(false, ah, false, bh, (short)0, c, false, false);
    c = __builtin_amdgcn_wmma_f32_16x16x32_f16(false, ah, false, bl, (short)0, c, false, false);
    c = __builtin_amdgcn_wmma_f32_16x16x32_f16(false, al, false, bh, (short)0, c, false, false);
  }

  // C/D layout: VGPR r -> M = r + 8*half, N = lane15
#pragma unroll
  for (int r = 0; r < 8; ++r)
    part[wave][half * 8 + r][lane15] = c[r];
  __syncthreads();

  // Reduce the 8 split-K partials; add bias.
  const int m = tid >> 4;
  const int n = tid & 15;
  float s = 0.0f;
#pragma unroll
  for (int w = 0; w < 8; ++w) s += part[w][m][n];
  if (n < NCLS) s += bias[n];
  part[0][m][n] = s;   // each (m,n) owned by exactly one thread -> safe
  __syncthreads();

  // Row softmax over the 10 valid classes.
  if (tid < 16) {
    float vals[NCLS];
    float mx = -3.4e38f;
    for (int j = 0; j < NCLS; ++j) {
      vals[j] = part[0][tid][j];
      mx = fmaxf(mx, vals[j]);
    }
    float sum = 0.0f;
    for (int j = 0; j < NCLS; ++j) {
      vals[j] = expf(vals[j] - mx);
      sum += vals[j];
    }
    const float inv = 1.0f / sum;
    for (int j = 0; j < NCLS; ++j)
      out[(size_t)(mtile * 16 + tid) * NCLS + j] = vals[j] * inv;
  }
}

// ---------------------------------------------------------------------------
extern "C" void kernel_launch(void* const* d_in, const int* in_sizes, int n_in,
                              void* d_out, int out_size, void* d_ws, size_t ws_size,
                              hipStream_t stream) {
  (void)in_sizes; (void)n_in; (void)out_size; (void)ws_size;
  const float* inputs = (const float*)d_in[0];  // (128, 4096, 2) f32
  const float* params = (const float*)d_in[1];  // (2, 1, 4) f32
  const float* W      = (const float*)d_in[2];  // (8190, 10) f32
  const float* b      = (const float*)d_in[3];  // (10,) f32
  float* out = (float*)d_out;                   // (128, 10) f32

  _Float16* Ghi = (_Float16*)d_ws;
  _Float16* Glo = Ghi + (size_t)B_DIM * K_DIM;
  _Float16* Whi = Glo + (size_t)B_DIM * K_DIM;
  _Float16* Wlo = Whi + (size_t)16 * K_DIM;     // total ~2.26 MB of d_ws

  qcnn_prep_g<<<(B_DIM * K_DIM) / 256, 256, 0, stream>>>(inputs, Ghi, Glo);
  qcnn_prep_w<<<(16 * K_DIM) / 256, 256, 0, stream>>>(W, params, Whi, Wlo);
  qcnn_gemm_softmax<<<8, 256, 0, stream>>>(Ghi, Glo, Whi, Wlo, b, out);
}
